// GSA_Attention_59579786330158
// MI455X (gfx1250) — matrix operand order repaired and checked by reading, not compiled, not run on hardware
//
#include <hip/hip_runtime.h>
#include <hip/hip_bf16.h>

// GSA attention for MI455X (gfx1250), wave32 + WMMA bf16 + async global->LDS.
// Pipeline: pre-convert x/w to bf16 -> qkv GEMM (SCALE folded into q) ->
// fused flash-attention (double-buffered async K/V tiles, gamma*A_phi bias,
// online softmax) -> output GEMM (f32 out).

#define BATCH 4
#define SEQ   2048
#define DM    512
#define NH    8
#define HD    64
#define SCALE_F 0.04419417382415922f   // 512^-0.5 (reference uses full dim)

typedef __attribute__((ext_vector_type(16))) __bf16 v16bf;
typedef __attribute__((ext_vector_type(8)))  float  v8f;
typedef __attribute__((ext_vector_type(2)))  __bf16 bf16x2;

union AB { v16bf v; unsigned int u[8]; };

static __device__ __forceinline__ unsigned short f2bf(float f) {
    return __builtin_bit_cast(unsigned short, (__bf16)f);
}
static __device__ __forceinline__ unsigned int pack2bf(float lo, float hi) {
    bf16x2 t; t.x = (__bf16)lo; t.y = (__bf16)hi;
    return __builtin_bit_cast(unsigned int, t);
}

// ---- CDNA5 async global->LDS copy (ASYNCcnt-tracked) ----------------------
static __device__ __forceinline__ void async_cp_b32(unsigned lds_off, unsigned g_off,
                                                    unsigned long long base) {
    asm volatile("global_load_async_to_lds_b32 %0, %1, %2"
                 :: "v"(lds_off), "v"(g_off), "s"(base) : "memory");
}
static __device__ __forceinline__ void wait_async_le8() {
    asm volatile("s_wait_asynccnt 0x8" ::: "memory");
}
static __device__ __forceinline__ void wait_async_0() {
    asm volatile("s_wait_asynccnt 0x0" ::: "memory");
}

// ---------------------------------------------------------------------------
// Kernel 0: convert f32 pairs -> packed bf16 (one-time precision shift).
// ---------------------------------------------------------------------------
__global__ __launch_bounds__(256)
void cvt_bf16_pairs(const float* __restrict__ src, unsigned int* __restrict__ dst,
                    int npairs)
{
    int i = blockIdx.x * 256 + threadIdx.x;
    if (i < npairs) dst[i] = pack2bf(src[2 * i], src[2 * i + 1]);
}

// ---------------------------------------------------------------------------
// Kernel 1: qkv = x @ w_qkv^T + b_qkv ; split into heads, bf16, SCALE into q.
// Inputs already bf16-packed; inner loop is pure loads + wmma.
// ---------------------------------------------------------------------------
__global__ __launch_bounds__(256)
void qkv_proj_kernel(const unsigned int* __restrict__ xbf,
                     const unsigned int* __restrict__ wbf,
                     const float* __restrict__ bias,
                     unsigned short* __restrict__ qh, unsigned short* __restrict__ kh,
                     unsigned short* __restrict__ vh)
{
    const int lane = threadIdx.x & 31;
    const int wave = threadIdx.x >> 5;
    const int g = lane >> 4, ln = lane & 15;
    const int rt = blockIdx.x;               // 512 row tiles (B*N/16)
    const int ct = blockIdx.y * 8 + wave;    // 96 col tiles (3D/16)
    const int row = rt * 16 + ln;
    const int col = ct * 16 + ln;
    const unsigned int* xr = xbf + (size_t)row * (DM / 2);
    const unsigned int* wr = wbf + (size_t)col * (DM / 2);

    v8f acc = {};
    for (int kk = 0; kk < DM / 32; ++kk) {
        AB a, b;
#pragma unroll
        for (int v = 0; v < 8; ++v) {
            int d = (v < 4) ? (kk * 32 + g * 8 + 2 * v)
                            : (kk * 32 + 16 + g * 8 + 2 * (v - 4));
            a.u[v] = xr[d >> 1];
            int k = (v < 4) ? (kk * 32 + g * 8 + 2 * v) : (kk * 32 + 16 + g * 8 + 2 * (v - 4));
            b.u[v] = wr[k >> 1];
        }
        acc = __builtin_amdgcn_wmma_f32_16x16x32_bf16(false, a.v, false, b.v,
                                                      (short)0, acc, false, false);
    }
    const int sel = col / DM;          // 0=q 1=k 2=v
    const int cm  = col % DM;
    const int h   = cm / HD;
    const int d   = cm % HD;
    const float bv = bias[col];
    unsigned short* dst = (sel == 0) ? qh : (sel == 1) ? kh : vh;
#pragma unroll
    for (int r = 0; r < 8; ++r) {
        int orow = rt * 16 + r + 8 * g;    // C/D layout: M = r + 8*(lane/16)
        int bb = orow >> 11;
        int nn = orow & (SEQ - 1);
        float val = acc[r] + bv;
        if (sel == 0) val *= SCALE_F;
        dst[((size_t)(bb * NH + h) * SEQ + nn) * HD + d] = f2bf(val);
    }
}

// ---------------------------------------------------------------------------
// Kernel 2: fused attention. Grid (qtile=16, h=8, b=4), 256 threads = 8 waves.
// K/V 32-key tiles double-buffered via global_load_async_to_lds_b32.
// ---------------------------------------------------------------------------
__global__ __launch_bounds__(256)
void attn_kernel(const unsigned short* __restrict__ qh,
                 const unsigned short* __restrict__ kh,
                 const unsigned short* __restrict__ vh,
                 const float* __restrict__ aphi,
                 const float* __restrict__ gamma_p,
                 unsigned short* __restrict__ oh)
{
    __shared__ unsigned int Kt[2][1024];     // 32 keys x 64 dims bf16, [k][d]
    __shared__ unsigned int Vt[2][1024];     // same natural layout
    __shared__ unsigned int Pbuf[8 * 256];   // per-wave 16x32 bf16 P scratch

    const int lane = threadIdx.x & 31;
    const int wave = threadIdx.x >> 5;
    const int g = lane >> 4, ln = lane & 15;
    const int b = blockIdx.z, h = blockIdx.y, qt = blockIdx.x;
    const float gamma = gamma_p[0];
    const int qbase = qt * 128 + wave * 16;
    const size_t headoff = (size_t)(b * NH + h) * SEQ * HD;
    unsigned short* Ps = (unsigned short*)Pbuf;

    const unsigned long long baseK = (unsigned long long)(size_t)(kh + headoff);
    const unsigned long long baseV = (unsigned long long)(size_t)(vh + headoff);
    const unsigned ldsK = (unsigned)(size_t)&Kt[0][0];
    const unsigned ldsV = (unsigned)(size_t)&Vt[0][0];

    // Q A-operands for this wave's 16 rows: two 32-wide K chunks (d=0..63)
    AB qa[2];
    {
        const unsigned int* qp32 =
            (const unsigned int*)(qh + headoff + (size_t)(qbase + ln) * HD);
#pragma unroll
        for (int c = 0; c < 2; ++c)
#pragma unroll
            for (int v = 0; v < 8; ++v) {
                int d = (v < 4) ? (c * 32 + g * 8 + 2 * v)
                                : (c * 32 + 16 + g * 8 + 2 * (v - 4));
                qa[c].u[v] = qp32[d >> 1];
            }
    }

    v8f acc[4] = {v8f{}, v8f{}, v8f{}, v8f{}};
    float mrow[8], lrow[8];
#pragma unroll
    for (int r = 0; r < 8; ++r) { mrow[r] = -3.0e38f; lrow[r] = 0.f; }

    const float* aprow = aphi + (size_t)b * SEQ * SEQ;
    const int NT = SEQ / 32;   // 64 key tiles

    // prologue: async-issue tile 0 into buffer 0 (8 instrs/wave)
#pragma unroll
    for (int i = 0; i < 4; ++i) {
        unsigned idx4 = ((unsigned)threadIdx.x + 256u * i) * 4u;
        async_cp_b32(ldsK + idx4, idx4, baseK);
        async_cp_b32(ldsV + idx4, idx4, baseV);
    }

    for (int ks = 0; ks < NT; ++ks) {
        const int kbase = ks * 32;
        const int buf = ks & 1;
        if (ks + 1 < NT) {
            // issue next tile while this one is consumed
            unsigned gofs = (unsigned)(ks + 1) * 4096u;
            unsigned lofs = (unsigned)((ks + 1) & 1) * 4096u;
#pragma unroll
            for (int i = 0; i < 4; ++i) {
                unsigned idx4 = ((unsigned)threadIdx.x + 256u * i) * 4u;
                async_cp_b32(ldsK + lofs + idx4, gofs + idx4, baseK);
                async_cp_b32(ldsV + lofs + idx4, gofs + idx4, baseV);
            }
            wait_async_le8();   // oldest 8 (tile ks) complete; in-order
        } else {
            wait_async_0();
        }
        __builtin_prefetch(aprow + (size_t)(qbase + ln) * SEQ + kbase + 32, 0, 1);
        __syncthreads();

        // -- S = q * k^T for two 16-key column tiles --
        v8f s0 = {}, s1 = {};
#pragma unroll
        for (int c = 0; c < 2; ++c) {
            AB b0, b1;
#pragma unroll
            for (int v = 0; v < 8; ++v) {
                int dp = (v < 4) ? (c * 32 + g * 8 + 2 * v) : (c * 32 + 16 + g * 8 + 2 * (v - 4)); // B K index = dim
                b0.u[v] = Kt[buf][ln * 32 + (dp >> 1)];       // keys 0..15
                b1.u[v] = Kt[buf][(16 + ln) * 32 + (dp >> 1)];// keys 16..31
            }
            s0 = __builtin_amdgcn_wmma_f32_16x16x32_bf16(false, qa[c].v, false, b0.v,
                                                         (short)0, s0, false, false);
            s1 = __builtin_amdgcn_wmma_f32_16x16x32_bf16(false, qa[c].v, false, b1.v,
                                                         (short)0, s1, false, false);
        }

        // -- bias + online softmax (row = r + 8*g, col = ln / 16+ln) --
#pragma unroll
        for (int r = 0; r < 8; ++r) {
            const int qrow = qbase + r + 8 * g;
            const float* ap = aprow + (size_t)qrow * SEQ + kbase;
            float v0 = s0[r] + gamma * ap[ln];
            float v1 = s1[r] + gamma * ap[16 + ln];
            float mx = fmaxf(v0, v1);
#pragma unroll
            for (int o = 1; o < 16; o <<= 1) mx = fmaxf(mx, __shfl_xor(mx, o, 32));
            float mnew = fmaxf(mrow[r], mx);
            float corr = __expf(mrow[r] - mnew);
            float p0 = __expf(v0 - mnew);
            float p1 = __expf(v1 - mnew);
            float rs = p0 + p1;
#pragma unroll
            for (int o = 1; o < 16; o <<= 1) rs += __shfl_xor(rs, o, 32);
            mrow[r] = mnew;
            lrow[r] = lrow[r] * corr + rs;
            acc[0][r] *= corr; acc[1][r] *= corr;
            acc[2][r] *= corr; acc[3][r] *= corr;
            int m = r + 8 * g;
            Ps[wave * 512 + m * 32 + ln]      = f2bf(p0);
            Ps[wave * 512 + m * 32 + 16 + ln] = f2bf(p1);
        }

        // -- out += P (16x32) * V (32x64), four 16-wide d tiles --
        AB pa;
#pragma unroll
        for (int v = 0; v < 8; ++v) {
            int kA = (v < 4) ? (g * 8 + 2 * v) : (16 + g * 8 + 2 * (v - 4));
            pa.u[v] = Pbuf[wave * 256 + ln * 16 + (kA >> 1)];
        }
        const unsigned short* Vs = (const unsigned short*)&Vt[buf][0];
#pragma unroll
        for (int j = 0; j < 4; ++j) {
            AB vb;
#pragma unroll
            for (int v = 0; v < 8; ++v) {
                int kp = (v < 4) ? (g * 8 + 2 * v) : (16 + g * 8 + 2 * (v - 4)); // key index pair
                unsigned lo = Vs[kp * HD + (j * 16 + ln)];
                unsigned hi = Vs[(kp + 1) * HD + (j * 16 + ln)];
                vb.u[v] = lo | (hi << 16);
            }
            acc[j] = __builtin_amdgcn_wmma_f32_16x16x32_bf16(false, pa.v, false, vb.v,
                                                             (short)0, acc[j], false, false);
        }
        __syncthreads();   // all waves done with buf before it is refilled
    }

    // -- normalize and store bf16 attention output [B,H,N,d] --
#pragma unroll
    for (int r = 0; r < 8; ++r) {
        float inv = 1.0f / lrow[r];
        int qrow = qbase + r + 8 * g;
        unsigned short* op = oh + headoff + (size_t)qrow * HD;
#pragma unroll
        for (int j = 0; j < 4; ++j)
            op[j * 16 + ln] = f2bf(acc[j][r] * inv);
    }
}

// ---------------------------------------------------------------------------
// Kernel 3: out = attn_out @ w_out^T + b_out (f32 result).
// ---------------------------------------------------------------------------
__global__ __launch_bounds__(256)
void out_proj_kernel(const unsigned short* __restrict__ oh,
                     const unsigned int* __restrict__ wbf,
                     const float* __restrict__ bias,
                     float* __restrict__ out)
{
    const int lane = threadIdx.x & 31;
    const int wave = threadIdx.x >> 5;
    const int g = lane >> 4, ln = lane & 15;
    const int rt = blockIdx.x;               // 512 row tiles
    const int ct = blockIdx.y * 8 + wave;    // 32 col tiles
    const int row = rt * 16 + ln;
    const int bb = row >> 11;
    const int nn = row & (SEQ - 1);
    const int col = ct * 16 + ln;
    const unsigned int* wr = wbf + (size_t)col * (DM / 2);

    v8f acc = {};
    for (int kk = 0; kk < DM / 32; ++kk) {
        AB a, b;
#pragma unroll
        for (int v = 0; v < 8; ++v) {
            int k = (v < 4) ? (kk * 32 + g * 8 + 2 * v)
                            : (kk * 32 + 16 + g * 8 + 2 * (v - 4));
            int hh = k >> 6, dd = k & 63;    // head-split bf16 input, pairs contiguous
            a.u[v] = *(const unsigned int*)(oh + ((size_t)(bb * NH + hh) * SEQ + nn) * HD + dd);
            int k2 = (v < 4) ? (kk * 32 + g * 8 + 2 * v) : (kk * 32 + 16 + g * 8 + 2 * (v - 4));
            b.u[v] = wr[k2 >> 1];
        }
        acc = __builtin_amdgcn_wmma_f32_16x16x32_bf16(false, a.v, false, b.v,
                                                      (short)0, acc, false, false);
    }
    const float bv = bias[col];
#pragma unroll
    for (int r = 0; r < 8; ++r)
        out[(size_t)(rt * 16 + r + 8 * g) * DM + col] = acc[r] + bv;
}

// ---------------------------------------------------------------------------
extern "C" void kernel_launch(void* const* d_in, const int* in_sizes, int n_in,
                              void* d_out, int out_size, void* d_ws, size_t ws_size,
                              hipStream_t stream)
{
    (void)in_sizes; (void)n_in; (void)out_size; (void)ws_size;
    const float* x     = (const float*)d_in[0];
    const float* aphi  = (const float*)d_in[1];
    const float* w_qkv = (const float*)d_in[2];
    const float* b_qkv = (const float*)d_in[3];
    const float* w_out = (const float*)d_in[4];
    const float* b_out = (const float*)d_in[5];
    const float* gamma = (const float*)d_in[6];
    float* out = (float*)d_out;

    const size_t per = (size_t)BATCH * NH * SEQ * HD;   // elems per head tensor
    unsigned short* qh = (unsigned short*)d_ws;
    unsigned short* kh = qh + per;
    unsigned short* vh = kh + per;
    unsigned short* oh = vh + per;
    unsigned int* xbf    = (unsigned int*)(oh + per);
    unsigned int* wqkvbf = xbf + (size_t)BATCH * SEQ * DM / 2;
    unsigned int* woutbf = wqkvbf + (size_t)3 * DM * DM / 2;

    const int xp = BATCH * SEQ * DM / 2;      // 2,097,152 pairs
    const int wqp = 3 * DM * DM / 2;          //   393,216 pairs
    const int wop = DM * DM / 2;              //   131,072 pairs
    cvt_bf16_pairs<<<dim3((xp + 255) / 256), 256, 0, stream>>>(x, xbf, xp);
    cvt_bf16_pairs<<<dim3((wqp + 255) / 256), 256, 0, stream>>>(w_qkv, wqkvbf, wqp);
    cvt_bf16_pairs<<<dim3((wop + 255) / 256), 256, 0, stream>>>(w_out, woutbf, wop);

    qkv_proj_kernel<<<dim3(512, 12), 256, 0, stream>>>(xbf, wqkvbf, b_qkv, qh, kh, vh);
    attn_kernel<<<dim3(SEQ / 128, NH, BATCH), 256, 0, stream>>>(qh, kh, vh, aphi, gamma, oh);
    out_proj_kernel<<<dim3(512, 4), 256, 0, stream>>>(oh, woutbf, b_out, out);
}
